// RayRoPE_DotProductAttention_84885733638602
// MI455X (gfx1250) — matrix operand order repaired and checked by reading, not compiled, not run on hardware
//
#include <hip/hip_runtime.h>

typedef __attribute__((ext_vector_type(16))) _Float16 v16h;
typedef __attribute__((ext_vector_type(8)))  _Float16 v8h;
typedef __attribute__((ext_vector_type(8)))  float    v8f;

#define B_ 2
#define H_ 16
#define S_ 2048
#define D_ 64

#define QT 128          // q rows per block (8 waves x 16)
#define KT 64           // keys per tile iteration
#define PAD 8
#define LDK (D_ + PAD)  // 72 f16 stride -> conflict-free ds_load_b128

union U16h { v16h v; v8h h[2]; _Float16 e[16]; };
union U8f  { v8f  v; float f[8]; };

__global__ __launch_bounds__(256)
void fa_rope_kernel(const float* __restrict__ Q, const float* __restrict__ K,
                    const float* __restrict__ V, const float* __restrict__ C,
                    const float* __restrict__ Sn, float* __restrict__ O)
{
    __shared__ _Float16 sK[KT][LDK];      // roped K, [key][dim]     (9216 B)
    __shared__ _Float16 sV[D_][LDK];      // V transposed [dim][key] (9216 B)
    __shared__ _Float16 sP[8][16][LDK];   // per-wave P  [row][key]  (18432 B)

    const int tid  = threadIdx.x;
    const int wave = tid >> 5;
    const int lane = tid & 31;
    const int ln   = lane & 15;   // column / row-within-16
    const int lh   = lane >> 4;   // lane half (selects K/M sub-range per ISA layout)

    const int b    = blockIdx.z;
    const int h    = blockIdx.y;
    const int qbase = blockIdx.x * QT + wave * 16;

    const size_t head = ((size_t)b * H_ + h) * S_;
    const float* qhead = Q + head * D_;
    const float* khead = K + head * D_;
    const float* vhead = V + head * D_;
    float*       ohead = O + head * D_;
    const float* chead = C  + (size_t)b * S_ * (D_/2);
    const float* shead = Sn + (size_t)b * S_ * (D_/2);

    // ---- per-wave Q: rope + fold (1/sqrt(D))*log2(e) + pack to WMMA A layout
    const float qscale = 0.125f * 1.44269504088896340736f;
    const int   qrow   = qbase + ln;
    const float* qp = qhead + (size_t)qrow * D_;
    const float* cq = chead + (size_t)qrow * (D_/2);
    const float* sq = shead + (size_t)qrow * (D_/2);

    U16h aq0, aq1;   // contraction chunks: dims [0,32) and [32,64)
    #pragma unroll
    for (int g = 0; g < 2; ++g) {
        #pragma unroll
        for (int j = 0; j < 8; ++j) {
            const int d  = lh * 16 + g * 8 + j;       // 0..31 (rope pair index)
            const float c  = cq[d], s = sq[d];
            const float x1 = qp[d], x2 = qp[d + 32];
            aq0.e[g * 8 + j] = (_Float16)((x1 * c - x2 * s) * qscale);
            aq1.e[g * 8 + j] = (_Float16)((x1 * s + x2 * c) * qscale);
        }
    }

    // ---- flash-attention state
    U8f acc[4];
    float m[8], l[8];
    #pragma unroll
    for (int o = 0; o < 4; ++o)
        #pragma unroll
        for (int r = 0; r < 8; ++r) acc[o].f[r] = 0.0f;
    #pragma unroll
    for (int r = 0; r < 8; ++r) { m[r] = -3.0e38f; l[r] = 0.0f; }

    for (int kt = 0; kt < S_ / KT; ++kt) {
        const int kb = kt * KT;

        // stage roped K -> sK : 2048 rope pairs, 8 per thread, coalesced
        #pragma unroll
        for (int p = 0; p < 8; ++p) {
            const int pe  = tid + p * 256;
            const int key = pe >> 5;
            const int dp  = pe & 31;
            const int ks  = kb + key;
            const float c  = chead[(size_t)ks * (D_/2) + dp];
            const float s  = shead[(size_t)ks * (D_/2) + dp];
            const float x1 = khead[(size_t)ks * D_ + dp];
            const float x2 = khead[(size_t)ks * D_ + dp + 32];
            sK[key][dp]      = (_Float16)(x1 * c - x2 * s);
            sK[key][dp + 32] = (_Float16)(x1 * s + x2 * c);
        }
        // stage V^T -> sV : coalesced global reads, strided f16 LDS writes
        #pragma unroll
        for (int p = 0; p < 16; ++p) {
            const int e   = tid + p * 256;
            const int key = e >> 6;
            const int d   = e & 63;
            sV[d][key] = (_Float16)vhead[(size_t)(kb + key) * D_ + d];
        }
        if (kt + 1 < S_ / KT) {   // global_prefetch_b8 on next tile
            __builtin_prefetch(khead + (size_t)(kb + KT) * D_ + tid * 16, 0, 0);
            __builtin_prefetch(vhead + (size_t)(kb + KT) * D_ + tid * 16, 0, 0);
        }
        __syncthreads();

        // ---- scores: S = Qr * Kr^T  (4 N-tiles x 2 K-chunks of 32)
        U8f st[4];
        #pragma unroll
        for (int t = 0; t < 4; ++t) {
            const _Float16* kp = &sK[t * 16 + ln][lh * 16];
            U16h bk0, bk1;
            bk0.h[0] = *(const v8h*)(kp);       bk0.h[1] = *(const v8h*)(kp + 8);
            bk1.h[0] = *(const v8h*)(kp + 32);  bk1.h[1] = *(const v8h*)(kp + 40);
            v8f z;
            #pragma unroll
            for (int r = 0; r < 8; ++r) z[r] = 0.0f;
            z = __builtin_amdgcn_wmma_f32_16x16x32_f16(false, aq0.v, false, bk0.v,
                                                       (short)0, z, false, false);
            z = __builtin_amdgcn_wmma_f32_16x16x32_f16(false, aq1.v, false, bk1.v,
                                                       (short)0, z, false, false);
            st[t].v = z;
        }

        // ---- online softmax (exp2 domain); rows live per lane-half
        float alpha[8];
        #pragma unroll
        for (int r = 0; r < 8; ++r) {
            float v0 = fmaxf(fmaxf(st[0].f[r], st[1].f[r]),
                             fmaxf(st[2].f[r], st[3].f[r]));
            v0 = fmaxf(v0, __shfl_xor(v0, 1, 32));
            v0 = fmaxf(v0, __shfl_xor(v0, 2, 32));
            v0 = fmaxf(v0, __shfl_xor(v0, 4, 32));
            v0 = fmaxf(v0, __shfl_xor(v0, 8, 32));
            const float mn = fmaxf(m[r], v0);
            alpha[r] = exp2f(m[r] - mn);
            m[r] = mn;
            float rs = 0.0f;
            #pragma unroll
            for (int t = 0; t < 4; ++t) {
                const float pv = exp2f(st[t].f[r] - mn);
                st[t].f[r] = pv;
                rs += pv;
            }
            rs += __shfl_xor(rs, 1, 32);
            rs += __shfl_xor(rs, 2, 32);
            rs += __shfl_xor(rs, 4, 32);
            rs += __shfl_xor(rs, 8, 32);
            l[r] = l[r] * alpha[r] + rs;
        }
        #pragma unroll
        for (int o = 0; o < 4; ++o)
            #pragma unroll
            for (int r = 0; r < 8; ++r) acc[o].f[r] *= alpha[r];

        // ---- P: D-layout -> A-layout via per-wave LDS slab (in-order DS)
        #pragma unroll
        for (int t = 0; t < 4; ++t)
            #pragma unroll
            for (int r = 0; r < 8; ++r)
                sP[wave][lh * 8 + r][t * 16 + ln] = (_Float16)st[t].f[r];

        // ---- O += P * V : contraction over 64 keys (2 chunks), 4 dim tiles
        #pragma unroll
        for (int c2 = 0; c2 < 2; ++c2) {
            const _Float16* pp = &sP[wave][ln][c2 * 32 + lh * 16];
            U16h ap;
            ap.h[0] = *(const v8h*)(pp);
            ap.h[1] = *(const v8h*)(pp + 8);
            #pragma unroll
            for (int o = 0; o < 4; ++o) {
                const _Float16* vp = &sV[o * 16 + ln][c2 * 32 + lh * 16];
                U16h bv;
                bv.h[0] = *(const v8h*)(vp);
                bv.h[1] = *(const v8h*)(vp + 8);
                acc[o].v = __builtin_amdgcn_wmma_f32_16x16x32_f16(
                    false, ap.v, false, bv.v, (short)0, acc[o].v, false, false);
            }
        }
        __syncthreads();
    }

    // ---- normalize + store (rows r / r+8 per lane half, cols = o*16+ln)
    #pragma unroll
    for (int o = 0; o < 4; ++o) {
        #pragma unroll
        for (int r = 0; r < 8; ++r) {
            const int row = qbase + lh * 8 + r;
            const int col = o * 16 + ln;
            ohead[(size_t)row * D_ + col] = acc[o].f[r] / l[r];
        }
    }
}

extern "C" void kernel_launch(void* const* d_in, const int* in_sizes, int n_in,
                              void* d_out, int out_size, void* d_ws, size_t ws_size,
                              hipStream_t stream) {
    const float* q = (const float*)d_in[0];
    const float* k = (const float*)d_in[1];
    const float* v = (const float*)d_in[2];
    const float* c = (const float*)d_in[3];
    const float* s = (const float*)d_in[4];
    float* o = (float*)d_out;
    dim3 grid(S_ / QT, H_, B_), block(256);
    hipLaunchKernelGGL(fa_rope_kernel, grid, block, 0, stream, q, k, v, c, s, o);
}
